// WindowAttention2D_89558658056370
// MI455X (gfx1250) — matrix operand order repaired and checked by reading, not compiled, not run on hardware
//
#include <hip/hip_runtime.h>
#include <hip/hip_bf16.h>

// ---------------------------------------------------------------------------
// Window attention (Swin-style), fully fused per window for MI455X (gfx1250).
//   B=4096 windows, N=64 tokens, DIM=512, HEADS=16, HEAD_DIM=32.
// All GEMMs run on v_wmma_f32_16x16x32_bf16 (wave32 WMMA).
// Wave w owns output-column slice [64w, 64w+64) in every projection, so each
// weight fragment is fetched exactly once per workgroup (min L2 traffic), and
// the attention phase (heads 2w, 2w+1) consumes only wave-local qh/K/V data.
// ---------------------------------------------------------------------------

typedef __attribute__((ext_vector_type(16))) __bf16 v16bf;
typedef __attribute__((ext_vector_type(8)))  float  v8f;

#define NTOK     64
#define DIMC     512
#define NHEADS   16
#define HEADD    32
#define NWIN     4096
#define NMASK    64
#define ATTN_SCALE 0.17677669529663689f   // 32^-0.5

// ws layout (bytes):
//   [0, 512K)        WqT  bf16 [512][512]   (n-major: WqT[n][k] = Wq[k][n])
//   [512K, 1536K)    WkvT bf16 [1024][512]
//   [1536K, 2048K)   WpT  bf16 [512][512]
//   [2048K, 2304K)   rpb  f32  [16][64][64]
#define WS_WKVT_OFF  262144     // elements (bf16)
#define WS_WPT_OFF   786432
#define WS_RPB_BYTE  2097152

union FragU { uint4 u[2]; v16bf v; };

// A-fragment (16x32 bf16, row-major source): per ISA table, lane = (m15, half);
// elements 0..7 -> K = h8..h8+7, elements 8..15 -> K = 16+h8..16+h8+7.
__device__ __forceinline__ v16bf ldA(const __bf16* p) {
    FragU f;
    f.u[0] = *(const uint4*)(p);
    f.u[1] = *(const uint4*)(p + 16);
    return f.v;
}
// B-fragment (32x16 bf16) loaded from the TRANSPOSED matrix (N-major rows):
// same K order as ldA: elements 0..7 -> K = h8..h8+7, 8..15 -> K = 16+h8..16+h8+7.
__device__ __forceinline__ v16bf ldB(const __bf16* p) {
    FragU f;
    f.u[0] = *(const uint4*)(p);
    f.u[1] = *(const uint4*)(p + 16);
    return f.v;
}

// ---------------------------------------------------------------------------
// Prep: transpose+convert weights to bf16, build relative-position bias table.
// ---------------------------------------------------------------------------
__global__ void winattn_prep_kernel(const float* __restrict__ Wq,
                                    const float* __restrict__ Wkv,
                                    const float* __restrict__ Wp,
                                    const float* __restrict__ bias_table,
                                    __bf16* __restrict__ wbuf,
                                    float* __restrict__ rpb) {
    int t = blockIdx.x * 256 + threadIdx.x;
    if (t < 262144) {                       // WqT
        int n = t >> 9, k = t & 511;
        wbuf[t] = (__bf16)Wq[k * 512 + n];
    } else if (t < 786432) {                // WkvT
        int t2 = t - 262144;
        int c = t2 >> 9, k = t2 & 511;
        wbuf[262144 + t2] = (__bf16)Wkv[k * 1024 + c];
    } else if (t < 1048576) {               // WpT
        int t2 = t - 786432;
        int n = t2 >> 9, k = t2 & 511;
        wbuf[786432 + t2] = (__bf16)Wp[k * 512 + n];
    } else if (t < 1114112) {               // rpb[h][m][n]
        int t2 = t - 1048576;
        int h = t2 >> 12;
        int m = (t2 >> 6) & 63;
        int n = t2 & 63;
        int ih = m >> 3, iw = m & 7, jh = n >> 3, jw = n & 7;
        int idx = (ih - jh + 7) * 15 + (iw - jw + 7);
        rpb[t2] = bias_table[idx * NHEADS + h];
    }
}

// ---------------------------------------------------------------------------
// Wave-level projection GEMM: [64 x 512] (LDS bf16) x WT^T [512 x *] + bias.
// Wave w: ALL 4 M-tiles x 4 N-tiles (columns [64w, 64w+64)) -> each weight
// fragment is read exactly once per workgroup.
// OUTMODE 0: bf16 row-major [64][512] to LDS
// OUTMODE 1: bf16 transposed [512][64] to LDS (for V)
// OUTMODE 2: fp32 row-major to global
// ---------------------------------------------------------------------------
template <int OUTMODE>
__device__ __forceinline__ void proj_gemm(const __bf16* __restrict__ sA,
                                          const __bf16* __restrict__ WT,
                                          const float* __restrict__ bias,
                                          __bf16* sOut, float* gOut,
                                          int lane, int waveId) {
    const int m15   = lane & 15;
    const int h8    = (lane >> 4) * 8;
    const int h16   = (lane >> 4) * 16;
    const int nBase = waveId * 64;          // this wave's 64-column slice

    v8f acc[4][4];                          // [mt][nt]
    v8f z = {};
#pragma unroll
    for (int mt = 0; mt < 4; ++mt)
#pragma unroll
        for (int nt = 0; nt < 4; ++nt) acc[mt][nt] = z;

#pragma unroll 1
    for (int kk = 0; kk < 16; ++kk) {
        v16bf a[4];
#pragma unroll
        for (int mt = 0; mt < 4; ++mt)
            a[mt] = ldA(sA + (mt * 16 + m15) * DIMC + kk * 32 + h8);
#pragma unroll
        for (int nt = 0; nt < 4; ++nt) {
            const __bf16* bp_ = WT + (nBase + nt * 16 + m15) * DIMC + kk * 32 + h8;
            v16bf b = ldB(bp_);
#pragma unroll
            for (int mt = 0; mt < 4; ++mt)
                acc[mt][nt] = __builtin_amdgcn_wmma_f32_16x16x32_bf16(
                    false, a[mt], false, b, (short)0, acc[mt][nt], false, false);
        }
    }

#pragma unroll
    for (int nt = 0; nt < 4; ++nt) {
        int n = nBase + nt * 16 + m15;
        float bb = bias[n];
#pragma unroll
        for (int mt = 0; mt < 4; ++mt) {
#pragma unroll
            for (int r = 0; r < 8; ++r) {
                int m = mt * 16 + r + h8;      // C layout: lanes16-31 -> rows +8
                float val = acc[mt][nt][r] + bb;
                if (OUTMODE == 0)      sOut[m * DIMC + n] = (__bf16)val;
                else if (OUTMODE == 1) sOut[n * NTOK + m] = (__bf16)val;
                else                   gOut[m * DIMC + n] = val;
            }
        }
    }
}

// ---------------------------------------------------------------------------
// Fused window-attention kernel. One block (8 wave32s, 256 KB LDS) per window.
// ---------------------------------------------------------------------------
__global__ void __launch_bounds__(256)
winattn_main_kernel(const float* __restrict__ q,
                    const float* __restrict__ kv,
                    const float* __restrict__ mask,
                    const float* __restrict__ bq,
                    const float* __restrict__ bkv,
                    const float* __restrict__ bp,
                    const __bf16* __restrict__ wbuf,
                    const float* __restrict__ rpb,
                    float* __restrict__ out) {
    extern __shared__ __bf16 sm[];
    __bf16* sIn = sm;            // 64x512 staging; later per-wave P slabs
    __bf16* sQh = sm + 32768;    // 64x512 qh (bf16)
    __bf16* sK  = sm + 65536;    // 64x512 k  (bf16); reused as x after attention
    __bf16* sVT = sm + 98304;    // 512x64 v transposed (bf16)

    const __bf16* WqT  = wbuf;
    const __bf16* WkvT = wbuf + WS_WKVT_OFF;
    const __bf16* WpT  = wbuf + WS_WPT_OFF;

    const int b      = blockIdx.x;
    const int tid    = threadIdx.x;
    const int lane   = tid & 31;
    const int waveId = tid >> 5;
    const int m15    = lane & 15;
    const int h8     = (lane >> 4) * 8;
    const int h16    = (lane >> 4) * 16;

    // ---- stage q -> bf16 LDS --------------------------------------------
    {
        const float4* src = (const float4*)(q + (size_t)b * (NTOK * DIMC));
#pragma unroll 4
        for (int i = 0; i < 32; ++i) {
            int j = i * 256 + tid;
            float4 v = src[j];
            __bf16* d = sIn + 4 * j;
            d[0] = (__bf16)v.x; d[1] = (__bf16)v.y;
            d[2] = (__bf16)v.z; d[3] = (__bf16)v.w;
        }
    }
    __syncthreads();

    // ---- qh = q @ Wq + bq ------------------------------------------------
    proj_gemm<0>(sIn, WqT, bq, sQh, nullptr, lane, waveId);
    __syncthreads();                       // before overwriting sIn with kv

    // ---- stage kv -> bf16 LDS -------------------------------------------
    {
        const float4* src = (const float4*)(kv + (size_t)b * (NTOK * DIMC));
#pragma unroll 4
        for (int i = 0; i < 32; ++i) {
            int j = i * 256 + tid;
            float4 v = src[j];
            __bf16* d = sIn + 4 * j;
            d[0] = (__bf16)v.x; d[1] = (__bf16)v.y;
            d[2] = (__bf16)v.z; d[3] = (__bf16)v.w;
        }
    }
    __syncthreads();

    // ---- k = kv @ Wkv[:, :512],  v = kv @ Wkv[:, 512:] (transposed) -----
    proj_gemm<0>(sIn, WkvT, bkv, sK, nullptr, lane, waveId);
    proj_gemm<1>(sIn, WkvT + 512 * DIMC, bkv + 512, sVT, nullptr, lane, waveId);
    __syncthreads();                       // sIn reused below as P slabs

    // ---- attention: wave w owns heads 2w, 2w+1 (exclusive column ranges) --
    {
        const float* maskw = mask + (size_t)(b & (NMASK - 1)) * (NTOK * NTOK);
        __bf16* sP = sIn + waveId * (NTOK * NTOK);   // private 8 KB P slab
        v8f z = {};

        for (int hh = 0; hh < 2; ++hh) {
            int h = waveId * 2 + hh;
            const float* rpbh = rpb + h * (NTOK * NTOK);

            // S = Q_h @ K_h^T  (64x64, single K=32 step)
            v16bf qa[4];
#pragma unroll
            for (int mt = 0; mt < 4; ++mt)
                qa[mt] = ldA(sQh + (mt * 16 + m15) * DIMC + h * HEADD + h8);
            v8f s[4][4];
#pragma unroll
            for (int nt = 0; nt < 4; ++nt) {
                v16bf kb = ldB(sK + (nt * 16 + m15) * DIMC + h * HEADD + h8);
#pragma unroll
                for (int mt = 0; mt < 4; ++mt)
                    s[mt][nt] = __builtin_amdgcn_wmma_f32_16x16x32_bf16(
                        false, qa[mt], false, kb, (short)0, z, false, false);
            }

            // scale + relative-position bias + window mask; row max
            float rmax[4][8];
#pragma unroll
            for (int mt = 0; mt < 4; ++mt)
#pragma unroll
                for (int r = 0; r < 8; ++r) rmax[mt][r] = -1e30f;
#pragma unroll
            for (int mt = 0; mt < 4; ++mt) {
#pragma unroll
                for (int nt = 0; nt < 4; ++nt) {
#pragma unroll
                    for (int r = 0; r < 8; ++r) {
                        int m = mt * 16 + r + h8;
                        int n = nt * 16 + m15;
                        float v = s[mt][nt][r] * ATTN_SCALE
                                + rpbh[m * NTOK + n] + maskw[m * NTOK + n];
                        s[mt][nt][r] = v;
                        rmax[mt][r] = fmaxf(rmax[mt][r], v);
                    }
                }
            }
            // reduce row-max across 16 lanes (xor 1,2,4,8 stays in each half)
#pragma unroll
            for (int mt = 0; mt < 4; ++mt)
#pragma unroll
                for (int r = 0; r < 8; ++r) {
                    float v = rmax[mt][r];
                    v = fmaxf(v, __shfl_xor(v, 1, 32));
                    v = fmaxf(v, __shfl_xor(v, 2, 32));
                    v = fmaxf(v, __shfl_xor(v, 4, 32));
                    v = fmaxf(v, __shfl_xor(v, 8, 32));
                    rmax[mt][r] = v;
                }
            // exp + row sum
            float rsum[4][8];
#pragma unroll
            for (int mt = 0; mt < 4; ++mt)
#pragma unroll
                for (int r = 0; r < 8; ++r) rsum[mt][r] = 0.f;
#pragma unroll
            for (int mt = 0; mt < 4; ++mt)
#pragma unroll
                for (int nt = 0; nt < 4; ++nt)
#pragma unroll
                    for (int r = 0; r < 8; ++r) {
                        float e = __expf(s[mt][nt][r] - rmax[mt][r]);
                        s[mt][nt][r] = e;
                        rsum[mt][r] += e;
                    }
#pragma unroll
            for (int mt = 0; mt < 4; ++mt)
#pragma unroll
                for (int r = 0; r < 8; ++r) {
                    float v = rsum[mt][r];
                    v += __shfl_xor(v, 1, 32);
                    v += __shfl_xor(v, 2, 32);
                    v += __shfl_xor(v, 4, 32);
                    v += __shfl_xor(v, 8, 32);
                    rsum[mt][r] = 1.0f / v;
                }
            // normalized P -> private LDS slab (bf16, row-major 64x64)
#pragma unroll
            for (int mt = 0; mt < 4; ++mt)
#pragma unroll
                for (int nt = 0; nt < 4; ++nt)
#pragma unroll
                    for (int r = 0; r < 8; ++r) {
                        int m = mt * 16 + r + h8;
                        int n = nt * 16 + m15;
                        sP[m * NTOK + n] = (__bf16)(s[mt][nt][r] * rsum[mt][r]);
                    }

            // O = P @ V_h  (64x32, K=64 -> two K=32 steps)
            v8f o[4][2];
#pragma unroll
            for (int mt = 0; mt < 4; ++mt) { o[mt][0] = z; o[mt][1] = z; }
#pragma unroll
            for (int kk = 0; kk < 2; ++kk) {
                v16bf pa[4];
#pragma unroll
                for (int mt = 0; mt < 4; ++mt)
                    pa[mt] = ldA(sP + (mt * 16 + m15) * NTOK + kk * 32 + h8);
#pragma unroll
                for (int nt = 0; nt < 2; ++nt) {
                    v16bf vb = ldB(sVT + (h * HEADD + nt * 16 + m15) * NTOK
                                   + kk * 32 + h8);
#pragma unroll
                    for (int mt = 0; mt < 4; ++mt)
                        o[mt][nt] = __builtin_amdgcn_wmma_f32_16x16x32_bf16(
                            false, pa[mt], false, vb, (short)0, o[mt][nt],
                            false, false);
                }
            }
            // x columns for head h overwrite this wave's (now dead) K columns
#pragma unroll
            for (int mt = 0; mt < 4; ++mt)
#pragma unroll
                for (int nt = 0; nt < 2; ++nt)
#pragma unroll
                    for (int r = 0; r < 8; ++r) {
                        int m = mt * 16 + r + h8;
                        int c = h * HEADD + nt * 16 + m15;
                        sK[m * DIMC + c] = (__bf16)o[mt][nt][r];
                    }
        }
    }
    __syncthreads();                       // all x columns visible

    // ---- out = x @ Wp + bp (fp32 to global) ------------------------------
    proj_gemm<2>(sK, WpT, bp, nullptr, out + (size_t)b * (NTOK * DIMC),
                 lane, waveId);
}

// ---------------------------------------------------------------------------
extern "C" void kernel_launch(void* const* d_in, const int* in_sizes, int n_in,
                              void* d_out, int out_size, void* d_ws, size_t ws_size,
                              hipStream_t stream) {
    const float* q    = (const float*)d_in[0];
    const float* kv   = (const float*)d_in[1];
    const float* mask = (const float*)d_in[2];
    const float* Wq   = (const float*)d_in[3];
    const float* bq   = (const float*)d_in[4];
    const float* Wkv  = (const float*)d_in[5];
    const float* bkv  = (const float*)d_in[6];
    const float* bt   = (const float*)d_in[7];
    const float* Wp   = (const float*)d_in[8];
    const float* bp   = (const float*)d_in[9];
    float* out = (float*)d_out;

    __bf16* wbuf = (__bf16*)d_ws;
    float*  rpb  = (float*)((char*)d_ws + WS_RPB_BYTE);

    // 1,114,112 conversion items -> 4352 blocks of 256
    winattn_prep_kernel<<<4352, 256, 0, stream>>>(Wq, Wkv, Wp, bt, wbuf, rpb);

    (void)hipFuncSetAttribute((const void*)winattn_main_kernel,
                              hipFuncAttributeMaxDynamicSharedMemorySize,
                              262144);
    winattn_main_kernel<<<NWIN, 256, 262144, stream>>>(
        q, kv, mask, bq, bkv, bp, wbuf, rpb, out);
}
